// Attention2_11871289606795
// MI455X (gfx1250) — matrix operand order repaired and checked by reading, not compiled, not run on hardware
//
#include <hip/hip_runtime.h>

// ---------------------------------------------------------------------------
// Problem dims
// ---------------------------------------------------------------------------
#define DIM_E 2048   // encoder dim (K of big GEMM)
#define DIM_A 1024   // attention dim (N of big GEMM)
#define DIM_H 1024   // hidden dim
#define NB    128    // batch
#define NP    196    // pixels
#define NROWS (NB * NP)   // 25088 = 392 * 64

typedef __attribute__((ext_vector_type(16))) __bf16 v16bf;
typedef __attribute__((ext_vector_type(8)))  __bf16 v8bf;
typedef __attribute__((ext_vector_type(8)))  float  v8f;

__device__ __forceinline__ unsigned short f2bf(float f) {
    unsigned int u = __float_as_uint(f);
    unsigned int r = u + 0x7FFFu + ((u >> 16) & 1u);   // round-to-nearest-even
    return (unsigned short)(r >> 16);
}

// ---------------------------------------------------------------------------
// Transpose + convert W_enc [2048,1024] fp32 -> W_encT bf16 [1024][2048]
// (4 MB, stays resident in the 192 MB L2 for the whole main GEMM)
// ---------------------------------------------------------------------------
__global__ __launch_bounds__(256) void k_transpose(const float* __restrict__ W_enc,
                                                   unsigned short* __restrict__ wT) {
    __shared__ unsigned short lds[32][36];   // padded
    const int k0 = blockIdx.x * 32;          // along E
    const int n0 = blockIdx.y * 32;          // along A
    const int t  = threadIdx.x;

    {   // load: 8 threads per k-row, each thread 4 consecutive n (float4, coalesced)
        const int k_l = t >> 3;
        const int n_q = (t & 7) * 4;
        float4 f = *(const float4*)(W_enc + (k0 + k_l) * DIM_A + n0 + n_q);
        lds[k_l][n_q + 0] = f2bf(f.x);
        lds[k_l][n_q + 1] = f2bf(f.y);
        lds[k_l][n_q + 2] = f2bf(f.z);
        lds[k_l][n_q + 3] = f2bf(f.w);
    }
    __syncthreads();
    {   // store: 8 threads per n-row, each thread 4 consecutive k (uint2, coalesced)
        const int n_l = t >> 3;
        const int k_q = (t & 7) * 4;
        uint2 o;
        o.x = (unsigned)lds[k_q + 0][n_l] | ((unsigned)lds[k_q + 1][n_l] << 16);
        o.y = (unsigned)lds[k_q + 2][n_l] | ((unsigned)lds[k_q + 3][n_l] << 16);
        *(uint2*)(wT + (size_t)(n0 + n_l) * DIM_E + k0 + k_q) = o;
    }
}

// ---------------------------------------------------------------------------
// bias2[b][n] = b_enc[n] + b_dec[n] + decoder_hidden[b,:] @ W_dec[:,n]
// ---------------------------------------------------------------------------
__global__ __launch_bounds__(256) void k_bias(const float* __restrict__ dh,
                                              const float* __restrict__ W_dec,
                                              const float* __restrict__ b_dec,
                                              const float* __restrict__ b_enc,
                                              float* __restrict__ bias2) {
    const int n = blockIdx.x * 256 + threadIdx.x;
    const int b = blockIdx.y;
    float acc = b_enc[n] + b_dec[n];
    const float* dhr = dh + b * DIM_H;
#pragma unroll 8
    for (int k = 0; k < DIM_H; ++k)
        acc = fmaf(dhr[k], W_dec[k * DIM_A + n], acc);
    bias2[b * DIM_A + n] = acc;
}

// ---------------------------------------------------------------------------
// Main fused kernel:
//   scores[r] = sum_n relu( enc[r,:] @ W_enc[:,n] + bias2[b(r),n] ) * W_full[n]
// 392 WGs x 256 threads (8 wave32). M=64 rows/WG, wave w owns 128 columns
// (8 n-tiles of 16). K stepped by 32; A tile double-buffered in LDS with
// register prefetch -> a single barrier per k-step.
// ---------------------------------------------------------------------------
__global__ __launch_bounds__(256) void k_main(const float* __restrict__ enc,
                                              const unsigned short* __restrict__ wT,
                                              const float* __restrict__ bias2,
                                              const float* __restrict__ W_full,
                                              float* __restrict__ scores) {
    __shared__ unsigned short Ald0[64][32];  // ping (4 KB)
    __shared__ unsigned short Ald1[64][32];  // pong (4 KB)
    __shared__ float sp_lds[8][64];          // per-wave partial row scores (2 KB)

    const int t    = threadIdx.x;
    const int w    = t >> 5;        // wave id 0..7
    const int lane = t & 31;
    const int ll   = lane & 15;     // N within tile / M within tile
    const int half = lane >> 4;     // lane-half selects K sub-range
    const int r0   = blockIdx.x * 64;

    v8f acc[4][8];
#pragma unroll
    for (int mt = 0; mt < 4; ++mt)
#pragma unroll
        for (int nc = 0; nc < 8; ++nc)
            acc[mt][nc] = (v8f){0.f, 0.f, 0.f, 0.f, 0.f, 0.f, 0.f, 0.f};

    float wf[8];
#pragma unroll
    for (int nc = 0; nc < 8; ++nc)
        wf[nc] = W_full[w * 128 + nc * 16 + ll];

    // staging: 4 threads per row, 8 fp32 each
    const int srow = t >> 2;
    const int sko  = (t & 3) * 8;
    const float* aptr = enc + (size_t)(r0 + srow) * DIM_E + sko;

    // prologue prefetch of first A tile into registers
    float4 f0 = *(const float4*)(aptr);
    float4 f1 = *(const float4*)(aptr + 4);

    // One k-step: store prefetched tile into Abuf, prefetch next tile,
    // single barrier, then 8 ds_load_b128 A-fragments + 16 global b128
    // B-fragments + 32 WMMAs.
    auto kstep = [&](unsigned short (*Abuf)[32], int k0, int knext) {
        {
            uint4 u;
            u.x = (unsigned)f2bf(f0.x) | ((unsigned)f2bf(f0.y) << 16);
            u.y = (unsigned)f2bf(f0.z) | ((unsigned)f2bf(f0.w) << 16);
            u.z = (unsigned)f2bf(f1.x) | ((unsigned)f2bf(f1.y) << 16);
            u.w = (unsigned)f2bf(f1.z) | ((unsigned)f2bf(f1.w) << 16);
            *(uint4*)&Abuf[srow][sko] = u;
        }
        // branch-free prefetch (wraps to 0 on the final step; harmless re-read)
        f0 = *(const float4*)(aptr + knext);
        f1 = *(const float4*)(aptr + knext + 4);
        __syncthreads();

        // A fragments: lane<16 holds K {0..15}, lane>=16 holds {16..31}
        v16bf afrag[4];
#pragma unroll
        for (int mt = 0; mt < 4; ++mt) {
            v8bf lo = *(const v8bf*)&Abuf[mt * 16 + ll][half * 16];
            v8bf hi = *(const v8bf*)&Abuf[mt * 16 + ll][half * 16 + 8];
#pragma unroll
            for (int i = 0; i < 8; ++i) {
                afrag[mt][i]     = lo[i];
                afrag[mt][i + 8] = hi[i];
            }
        }
        // B fragments from L2-resident W_encT: 32 contiguous bytes per lane
#pragma unroll
        for (int nc = 0; nc < 8; ++nc) {
            const int n = w * 128 + nc * 16 + ll;
            v16bf bfrag = *(const v16bf*)(wT + (size_t)n * DIM_E + k0 + half * 16);
#pragma unroll
            for (int mt = 0; mt < 4; ++mt)
                acc[mt][nc] = __builtin_amdgcn_wmma_f32_16x16x32_bf16(
                    false, afrag[mt], false, bfrag, (short)0, acc[mt][nc], false, false);
        }
    };

#pragma unroll 1
    for (int k0 = 0; k0 < DIM_E; k0 += 64) {
        kstep(Ald0, k0,      k0 + 32);
        kstep(Ald1, k0 + 32, (k0 + 64) & (DIM_E - 1));
    }

    // ---- fused epilogue: bias + relu + dot(W_full) + per-row reduce
#pragma unroll
    for (int mt = 0; mt < 4; ++mt) {
        float sp[8];
#pragma unroll
        for (int v = 0; v < 8; ++v) sp[v] = 0.f;
#pragma unroll
        for (int nc = 0; nc < 8; ++nc) {
            const int n = w * 128 + nc * 16 + ll;
#pragma unroll
            for (int v = 0; v < 8; ++v) {
                const int r = r0 + mt * 16 + half * 8 + v;   // global row
                const int b = r / NP;                        // batch of this row
                float x = acc[mt][nc][v] + bias2[b * DIM_A + n];
                x = fmaxf(x, 0.f);
                sp[v] = fmaf(x, wf[nc], sp[v]);
            }
        }
#pragma unroll
        for (int v = 0; v < 8; ++v) {
            float s = sp[v];
            s += __shfl_xor(s, 1, 32);
            s += __shfl_xor(s, 2, 32);
            s += __shfl_xor(s, 4, 32);
            s += __shfl_xor(s, 8, 32);   // sum over 16 lanes within each half
            if (ll == 0)
                sp_lds[w][mt * 16 + half * 8 + v] = s;
        }
    }
    __syncthreads();
    // deterministic cross-wave reduction (no float atomics)
    if (t < 64) {
        float s = 0.f;
#pragma unroll
        for (int ww = 0; ww < 8; ++ww) s += sp_lds[ww][t];
        scores[r0 + t] = s;
    }
}

// ---------------------------------------------------------------------------
// Softmax over P per batch (b_full dropped: softmax is shift-invariant)
// ---------------------------------------------------------------------------
__global__ __launch_bounds__(256) void k_softmax(const float* __restrict__ scores,
                                                 float* __restrict__ alpha) {
    __shared__ float red[256];
    const int b = blockIdx.x;
    const int t = threadIdx.x;
    float x = (t < NP) ? scores[b * NP + t] : -3.4e38f;
    red[t] = x;
    __syncthreads();
    for (int s = 128; s > 0; s >>= 1) {
        if (t < s) red[t] = fmaxf(red[t], red[t + s]);
        __syncthreads();
    }
    const float mx = red[0];
    __syncthreads();
    float e = (t < NP) ? __expf(x - mx) : 0.f;
    red[t] = e;
    __syncthreads();
    for (int s = 128; s > 0; s >>= 1) {
        if (t < s) red[t] += red[t + s];
        __syncthreads();
    }
    const float inv = 1.f / red[0];
    if (t < NP) alpha[b * NP + t] = e * inv;
}

// ---------------------------------------------------------------------------
// context[b,e] = sum_p enc[b,p,e] * alpha[b,p]   (memory bound, coalesced on e)
// ---------------------------------------------------------------------------
__global__ __launch_bounds__(256) void k_context(const float* __restrict__ enc,
                                                 const float* __restrict__ alpha,
                                                 float* __restrict__ ctx) {
    __shared__ float sal[NP];
    const int b = blockIdx.y;
    const int e = blockIdx.x * 256 + threadIdx.x;
    if (threadIdx.x < NP) sal[threadIdx.x] = alpha[b * NP + threadIdx.x];
    __syncthreads();
    const float* base = enc + (size_t)b * NP * DIM_E + e;
    float s = 0.f;
#pragma unroll 4
    for (int p = 0; p < NP; ++p)
        s = fmaf(base[(size_t)p * DIM_E], sal[p], s);
    ctx[b * DIM_E + e] = s;
}

// ---------------------------------------------------------------------------
extern "C" void kernel_launch(void* const* d_in, const int* in_sizes, int n_in,
                              void* d_out, int out_size, void* d_ws, size_t ws_size,
                              hipStream_t stream) {
    const float* enc    = (const float*)d_in[0];  // [128,196,2048]
    const float* dh     = (const float*)d_in[1];  // [128,1024]
    const float* W_enc  = (const float*)d_in[2];  // [2048,1024]
    const float* b_enc  = (const float*)d_in[3];  // [1024]
    const float* W_dec  = (const float*)d_in[4];  // [1024,1024]
    const float* b_dec  = (const float*)d_in[5];  // [1024]
    const float* W_full = (const float*)d_in[6];  // [1024]
    // d_in[7] = b_full : unused (softmax is shift-invariant, scores not output)

    // workspace layout
    char* ws = (char*)d_ws;
    unsigned short* wT     = (unsigned short*)ws;                       // 4 MB
    float*          bias2  = (float*)(ws + (size_t)DIM_A * DIM_E * 2);  // 512 KB
    float*          scores = bias2 + (size_t)NB * DIM_A;                // ~100 KB

    float* ctx_out   = (float*)d_out;                 // [128,2048]
    float* alpha_out = ctx_out + (size_t)NB * DIM_E;  // [128,196]

    k_transpose<<<dim3(DIM_E / 32, DIM_A / 32), 256, 0, stream>>>(W_enc, wT);
    k_bias<<<dim3(DIM_A / 256, NB), 256, 0, stream>>>(dh, W_dec, b_dec, b_enc, bias2);
    k_main<<<NROWS / 64, 256, 0, stream>>>(enc, wT, bias2, W_full, scores);
    k_softmax<<<NB, 256, 0, stream>>>(scores, alpha_out);
    k_context<<<dim3(DIM_E / 256, NB), 256, 0, stream>>>(enc, alpha_out, ctx_out);
}
